// CausalSelfAttention_34626026341061
// MI455X (gfx1250) — matrix operand order repaired and checked by reading, not compiled, not run on hardware
//
#include <hip/hip_runtime.h>
#include <hip/hip_bf16.h>
#include <cstdint>

typedef __bf16 bf16_t;
typedef __attribute__((ext_vector_type(16))) __bf16 v16bf;
typedef __attribute__((ext_vector_type(8)))  float  v8f;

#ifndef __has_builtin
#define __has_builtin(x) 0
#endif
#if __has_builtin(__builtin_amdgcn_global_load_async_to_lds_b128)
#define HAVE_ASYNC_LDS 1
#else
#define HAVE_ASYNC_LDS 0
#endif

// Pointer types matching the async builtin's signature: int4-vector pointee,
// AS1 (global) and AS3 (LDS).
typedef int v4i_vs __attribute__((vector_size(16)));
typedef __attribute__((address_space(1))) v4i_vs* as1_b128_t;
typedef __attribute__((address_space(3))) v4i_vs* as3_b128_t;

union FragU { uint4 u[2]; v16bf v; };

// Build a 16-element bf16 fragment from two 16-byte chunks (LDS b128 loads).
__device__ __forceinline__ v16bf ld_frag(const bf16_t* p0, const bf16_t* p1) {
  FragU f;
  f.u[0] = *(const uint4*)p0;
  f.u[1] = *(const uint4*)p1;
  return f.v;
}

__device__ __forceinline__ v8f wmma_bf16(v16bf a, v16bf b, v8f c) {
  return __builtin_amdgcn_wmma_f32_16x16x32_bf16(
      false, a, false, b, (short)0, c, false, false);
}

// Async global->LDS 16-byte copy (CDNA5 GLOBAL_LOAD_ASYNC_TO_LDS_B128),
// falling back to a through-VGPR copy if the builtin is unavailable.
// Integer round-trips give correct AS1/AS3 addresses on amdgcn (generic
// global == AS1; generic LDS truncates to the 32-bit LDS offset).
__device__ __forceinline__ void async_copy_b128(const bf16_t* g, bf16_t* l) {
#if HAVE_ASYNC_LDS
  __builtin_amdgcn_global_load_async_to_lds_b128(
      (as1_b128_t)(uintptr_t)g,
      (as3_b128_t)(unsigned int)(uintptr_t)l,
      0, 0);
#else
  *(uint4*)l = *(const uint4*)g;
#endif
}

__device__ __forceinline__ void async_wait() {
#if HAVE_ASYNC_LDS
  asm volatile("s_wait_asynccnt 0x0" ::: "memory");
#endif
}

// ---------------------------------------------------------------------------
// fp32 -> bf16 elementwise convert (n is a multiple of 4)
// ---------------------------------------------------------------------------
__global__ __launch_bounds__(256) void f32_to_bf16_kernel(
    const float* __restrict__ in, bf16_t* __restrict__ out, int n4) {
  int i = blockIdx.x * blockDim.x + threadIdx.x;
  if (i < n4) {
    float4 v = ((const float4*)in)[i];
    union { bf16_t h[4]; uint2 u; } o;
    o.h[0] = (bf16_t)v.x; o.h[1] = (bf16_t)v.y;
    o.h[2] = (bf16_t)v.z; o.h[3] = (bf16_t)v.w;
    ((uint2*)out)[i] = o.u;
  }
}

// ---------------------------------------------------------------------------
// bf16 WMMA GEMM: C[M,N] = A[M,K] * B[K,N].  All strides compile-time.
// 256 threads = 8 waves; workgroup tile 128x128; wave tile 32x64.
// Double-buffered, software-pipelined: async A-copies + B global loads for
// tile t+1 are issued before the 16 WMMAs of tile t; one barrier per round.
// ---------------------------------------------------------------------------
template <typename OutT, int Kc, int LDA, int LDB, int LDC>
__global__ __launch_bounds__(256) void gemm_bf16_wmma(
    const bf16_t* __restrict__ A, const bf16_t* __restrict__ B,
    OutT* __restrict__ Cout) {
  __shared__ bf16_t As[2][128 * 64];
  __shared__ bf16_t Bt[2][128 * 64];

  const int tid  = threadIdx.x;
  const int lane = tid & 31, wv = tid >> 5;
  const int g = lane >> 4, ln = lane & 15;
  const int wgM = blockIdx.y * 128, wgN = blockIdx.x * 128;
  const int m0 = (wv & 3) * 32;   // wave row offset within tile
  const int n0 = (wv >> 2) * 64;  // wave col offset within tile

  v8f acc[2][4];
#pragma unroll
  for (int a = 0; a < 2; ++a)
#pragma unroll
    for (int b = 0; b < 4; ++b)
      acc[a][b] = (v8f){0.f, 0.f, 0.f, 0.f, 0.f, 0.f, 0.f, 0.f};

  const int ar = tid >> 1, ac = (tid & 1) * 32;   // A staging: row, k-col
  const int bk = tid >> 2, bn = (tid & 3) * 32;   // B staging: k-row, n-col

  const bf16_t* arow = A + (size_t)(wgM + ar) * LDA + ac;  // + k0
  const bf16_t* brow = B + (size_t)bk * LDB + (wgN + bn);  // + k0*LDB

  // Prologue: stage tile 0 into buffer 0.
  {
    bf16_t* adst = &As[0][ar * 64 + ac];
#pragma unroll
    for (int i = 0; i < 4; ++i) async_copy_b128(arow + i * 8, adst + i * 8);
    bf16_t tmp[32];
#pragma unroll
    for (int i = 0; i < 4; ++i)
      *(uint4*)(&tmp[i * 8]) = *(const uint4*)(brow + i * 8);
#pragma unroll
    for (int i = 0; i < 32; ++i) Bt[0][(bn + i) * 64 + bk] = tmp[i];
  }
  async_wait();
  __syncthreads();

  for (int k0 = 0; k0 < Kc; k0 += 64) {
    const int cur = (k0 >> 6) & 1, nxt = cur ^ 1;
    const bool hasNext = (k0 + 64) < Kc;

    // Prefetch tile t+1: B rows into VGPRs, A rows async into LDS.
    bf16_t tmp[32];
    if (hasNext) {
      const bf16_t* bsrc = brow + (size_t)(k0 + 64) * LDB;
#pragma unroll
      for (int i = 0; i < 4; ++i)
        *(uint4*)(&tmp[i * 8]) = *(const uint4*)(bsrc + i * 8);
      const bf16_t* asrc = arow + (k0 + 64);
      bf16_t* adst = &As[nxt][ar * 64 + ac];
#pragma unroll
      for (int i = 0; i < 4; ++i) async_copy_b128(asrc + i * 8, adst + i * 8);
    }

    // Compute on tile t (hides the prefetch latency behind WMMA issue).
#pragma unroll
    for (int s = 0; s < 64; s += 32) {
      v16bf af[2], bfr[4];
#pragma unroll
      for (int mi = 0; mi < 2; ++mi) {
        const bf16_t* r = &As[cur][(m0 + mi * 16 + ln) * 64 + s];
        af[mi] = ld_frag(r + g * 16, r + g * 16 + 8);  // 16 contiguous K
      }
#pragma unroll
      for (int ni = 0; ni < 4; ++ni) {
        const bf16_t* r = &Bt[cur][(n0 + ni * 16 + ln) * 64 + s + g * 16];
        bfr[ni] = ld_frag(r, r + 8);                  // 16 contiguous K
      }
#pragma unroll
      for (int mi = 0; mi < 2; ++mi)
#pragma unroll
        for (int ni = 0; ni < 4; ++ni)
          acc[mi][ni] = wmma_bf16(af[mi], bfr[ni], acc[mi][ni]);
    }

    // Transposed B store for tile t+1 (into the other buffer).
    if (hasNext) {
#pragma unroll
      for (int i = 0; i < 32; ++i) Bt[nxt][(bn + i) * 64 + bk] = tmp[i];
    }
    async_wait();
    __syncthreads();
  }

  // Epilogue: one per-lane base pointer, all offsets compile-time immediates.
  OutT* cp = Cout + (size_t)(wgM + m0 + g * 8) * LDC + (wgN + n0 + ln);
#pragma unroll
  for (int mi = 0; mi < 2; ++mi)
#pragma unroll
    for (int i = 0; i < 8; ++i)
#pragma unroll
      for (int ni = 0; ni < 4; ++ni)
        cp[(size_t)(mi * 16 + i) * LDC + ni * 16] = (OutT)acc[mi][ni][i];
}

// ---------------------------------------------------------------------------
// Flash attention (causal), bf16 WMMA, fp32 softmax state.
// Grid: (B*H, T/64). 128 threads = 4 waves; each wave owns 16 query rows.
// qkv layout: [B*T, 3C]; q at col h*128, k at C+h*128, v at 2C+h*128.
// Double-buffered K/V staging, pipelined against S/softmax/O compute.
// ---------------------------------------------------------------------------
__global__ __launch_bounds__(128) void flash_attn_wmma(
    const bf16_t* __restrict__ qkv, bf16_t* __restrict__ outb, float scale) {
  constexpr int Cc = 2048, Tc = 2048, HD = 128, LD = 3 * Cc;
  __shared__ bf16_t Ks[2][32 * 128];   // K tile, row-major [key][d]
  __shared__ bf16_t Vt[2][128 * 32];   // V tile transposed [d][key]
  __shared__ bf16_t Ps[4][16 * 32];    // per-wave P patch [q][key]

  const int tid  = threadIdx.x;
  const int lane = tid & 31, wv = tid >> 5;
  const int g = lane >> 4, ln = lane & 15;
  const int b = blockIdx.x >> 4, h = blockIdx.x & 15;
  const int qblk = blockIdx.y * 64;
  const int q0 = qblk + wv * 16;

  const bf16_t* qbase = qkv + (size_t)(b * Tc) * LD + h * HD;
  const bf16_t* kbase = qbase + Cc;
  const bf16_t* vbase = qbase + 2 * Cc;

  // Q fragments for d = 0..127 in four 32-wide chunks (held in VGPRs).
  v16bf qf[4];
  {
    const bf16_t* qrow = qbase + (size_t)(q0 + ln) * LD + g * 16;
#pragma unroll
    for (int dc = 0; dc < 4; ++dc)
      qf[dc] = ld_frag(qrow + dc * 32, qrow + dc * 32 + 8);
  }

  v8f of[8];
#pragma unroll
  for (int dc = 0; dc < 8; ++dc)
    of[dc] = (v8f){0.f, 0.f, 0.f, 0.f, 0.f, 0.f, 0.f, 0.f};
  float mrow[8], lrow[8];
#pragma unroll
  for (int i = 0; i < 8; ++i) { mrow[i] = -1e30f; lrow[i] = 0.f; }

  const int kk = tid >> 2, dd = (tid & 3) * 32;  // staging coords
  const int kend = qblk + 64;                    // causal bound for this block

  // Prologue: stage key tile 0 into buffer 0.
  {
    const bf16_t* ks = kbase + (size_t)kk * LD + dd;
    bf16_t* kd = &Ks[0][kk * 128 + dd];
#pragma unroll
    for (int i = 0; i < 4; ++i) async_copy_b128(ks + i * 8, kd + i * 8);
    const bf16_t* vs = vbase + (size_t)kk * LD + dd;
    bf16_t tmp[32];
#pragma unroll
    for (int i = 0; i < 4; ++i)
      *(uint4*)(&tmp[i * 8]) = *(const uint4*)(vs + i * 8);
#pragma unroll
    for (int i = 0; i < 32; ++i) Vt[0][(dd + i) * 32 + kk] = tmp[i];
  }
  async_wait();
  __syncthreads();

  for (int key0 = 0; key0 < kend; key0 += 32) {
    const int cur = (key0 >> 5) & 1, nxt = cur ^ 1;
    const bool hasNext = (key0 + 32) < kend;

    // Prefetch next key tile: K async to LDS, V rows into VGPRs.
    bf16_t vtmp[32];
    if (hasNext) {
      const bf16_t* ks = kbase + (size_t)(key0 + 32 + kk) * LD + dd;
      bf16_t* kd = &Ks[nxt][kk * 128 + dd];
#pragma unroll
      for (int i = 0; i < 4; ++i) async_copy_b128(ks + i * 8, kd + i * 8);
      const bf16_t* vs = vbase + (size_t)(key0 + 32 + kk) * LD + dd;
#pragma unroll
      for (int i = 0; i < 4; ++i)
        *(uint4*)(&vtmp[i * 8]) = *(const uint4*)(vs + i * 8);
    }

    // S = Q * K^T for two 16-key sub-tiles.
    v8f s0 = (v8f){0.f, 0.f, 0.f, 0.f, 0.f, 0.f, 0.f, 0.f};
    v8f s1 = s0;
#pragma unroll
    for (int dc = 0; dc < 4; ++dc) {
      const bf16_t* r0 = &Ks[cur][ln * 128 + dc * 32 + g * 16];
      const bf16_t* r1 = &Ks[cur][(16 + ln) * 128 + dc * 32 + g * 16];
      s0 = wmma_bf16(qf[dc], ld_frag(r0, r0 + 8), s0);
      s1 = wmma_bf16(qf[dc], ld_frag(r1, r1 + 8), s1);
    }

    // Online softmax update (rows g*8+i, cols across lanes of half-wave).
    const int qrow = q0 + g * 8;
#pragma unroll
    for (int i = 0; i < 8; ++i) {
      float v0 = s0[i] * scale;
      float v1 = s1[i] * scale;
      if (key0 + ln > qrow + i)      v0 = -1e30f;   // causal mask
      if (key0 + 16 + ln > qrow + i) v1 = -1e30f;
      float rm = fmaxf(v0, v1);
#pragma unroll
      for (int m = 1; m < 16; m <<= 1) rm = fmaxf(rm, __shfl_xor(rm, m, 32));
      float mnew = fmaxf(mrow[i], rm);
      float p0 = __expf(v0 - mnew);
      float p1 = __expf(v1 - mnew);
      float rs = p0 + p1;
#pragma unroll
      for (int m = 1; m < 16; m <<= 1) rs += __shfl_xor(rs, m, 32);
      float alpha = __expf(mrow[i] - mnew);
      lrow[i] = lrow[i] * alpha + rs;
      mrow[i] = mnew;
#pragma unroll
      for (int dc = 0; dc < 8; ++dc) of[dc][i] *= alpha;
      // C-layout -> LDS (re-read below in A-layout). Per-wave region.
      Ps[wv][(g * 8 + i) * 32 + ln]      = (bf16_t)p0;
      Ps[wv][(g * 8 + i) * 32 + 16 + ln] = (bf16_t)p1;
    }
    // Same-wave LDS ordering: all P stores complete before fragment reload.
    asm volatile("s_wait_dscnt 0x0" ::: "memory");

    const bf16_t* pr = &Ps[wv][ln * 32 + g * 16];
    v16bf pf = ld_frag(pr, pr + 8);

    // O += P * V, eight 16-wide d chunks.
#pragma unroll
    for (int dc = 0; dc < 8; ++dc) {
      const bf16_t* vr = &Vt[cur][(dc * 16 + ln) * 32 + g * 16];
      of[dc] = wmma_bf16(pf, ld_frag(vr, vr + 8), of[dc]);
    }

    // Transposed V store for the next tile (other buffer).
    if (hasNext) {
#pragma unroll
      for (int i = 0; i < 32; ++i) Vt[nxt][(dd + i) * 32 + kk] = vtmp[i];
    }
    async_wait();
    __syncthreads();
  }

  // Normalize and store merged-head output as bf16 [B*T, C].
  bf16_t* op = outb + (size_t)(b * Tc + q0 + g * 8) * Cc + h * HD + ln;
#pragma unroll
  for (int i = 0; i < 8; ++i) {
    float inv = 1.f / lrow[i];
#pragma unroll
    for (int dc = 0; dc < 8; ++dc)
      op[(size_t)i * Cc + dc * 16] = (bf16_t)(of[dc][i] * inv);
  }
}

// ---------------------------------------------------------------------------
// Host-side launcher
// ---------------------------------------------------------------------------
extern "C" void kernel_launch(void* const* d_in, const int* in_sizes, int n_in,
                              void* d_out, int out_size, void* d_ws, size_t ws_size,
                              hipStream_t stream) {
  (void)in_sizes; (void)n_in; (void)out_size; (void)ws_size;
  constexpr int Bb = 4, Tt = 2048, Cc = 2048;

  const float* x      = (const float*)d_in[0];
  const float* w_qkv  = (const float*)d_in[1];
  const float* w_proj = (const float*)d_in[2];
  float* out = (float*)d_out;

  char* ws = (char*)d_ws;
  size_t off = 0;
  auto walloc = [&](size_t elems) {
    bf16_t* p = (bf16_t*)(ws + off);
    off += (elems * sizeof(bf16_t) + 255) & ~(size_t)255;
    return p;
  };
  const size_t nX = (size_t)Bb * Tt * Cc;
  bf16_t* xb   = walloc(nX);                       // 32 MB
  bf16_t* wqb  = walloc((size_t)Cc * 3 * Cc);      // 24 MB
  bf16_t* wpb  = walloc((size_t)Cc * Cc);          //  8 MB
  bf16_t* qkvb = walloc((size_t)Bb * Tt * 3 * Cc); // 96 MB
  bf16_t* attb = walloc(nX);                       // 32 MB

  auto conv = [&](const float* src, bf16_t* dst, size_t n) {
    int n4 = (int)(n / 4);
    f32_to_bf16_kernel<<<(n4 + 255) / 256, 256, 0, stream>>>(src, dst, n4);
  };
  conv(x, xb, nX);
  conv(w_qkv, wqb, (size_t)Cc * 3 * Cc);
  conv(w_proj, wpb, (size_t)Cc * Cc);

  // qkv = x @ w_qkv : [8192,2048] x [2048,6144] -> bf16 [8192,6144]
  gemm_bf16_wmma<bf16_t, 2048, 2048, 6144, 6144>
      <<<dim3(3 * Cc / 128, Bb * Tt / 128), 256, 0, stream>>>(xb, wqb, qkvb);

  // causal flash attention -> bf16 [8192,2048]
  flash_attn_wmma<<<dim3(Bb * 16, Tt / 64), 128, 0, stream>>>(
      qkvb, attb, 0.088388347648318447f);

  // out = attn @ w_proj : [8192,2048] x [2048,2048] -> fp32 d_out
  gemm_bf16_wmma<float, 2048, 2048, 2048, 2048>
      <<<dim3(Cc / 128, Bb * Tt / 128), 256, 0, stream>>>(attb, wpb, out);
}
